// VQVAE_11802570130397
// MI455X (gfx1250) — matrix operand order repaired and checked by reading, not compiled, not run on hardware
//
#include <hip/hip_runtime.h>
#include <hip/hip_bf16.h>

// ---------------------------------------------------------------------------
// VQ-VAE forward for gfx1250 (MI455X).
// All conv / deconv / VQ-distance GEMMs run through v_wmma_f32_16x16x32_f16
// (f16 operands, f32 accumulation). Activations live in f16 (bandwidth-bound:
// ~26 GFLOP vs ~0.2 GB). Implicit-GEMM: 32(M=Cout) x 128(N=B*Ho*Wo) tile per
// 256-thread block (8 waves), K chunked by 32. Weight fragments are loaded
// straight from global (b128 clauses); im2col taps are precomputed per thread
// (spatial dims are powers of two -> shift/mask addressing, clamped offsets),
// spatial validity is a chunk-invariant 0/1 f16 vector applied with
// v_pk_mul_f16 (out-of-range K columns hit zero-padded weight rows, so only
// spatial masking is needed), and the gather is software-pipelined one
// K-chunk ahead so its latency hides behind the barrier + WMMA.
// ---------------------------------------------------------------------------

typedef __attribute__((ext_vector_type(16))) _Float16 v16h;
typedef __attribute__((ext_vector_type(8)))  _Float16 v8h;
typedef __attribute__((ext_vector_type(8)))  float    v8f;

#define TILE_M 32
#define TILE_N 128
#define KC     32

// ---------------------------------------------------------------------------
// Pack helpers
// ---------------------------------------------------------------------------
__global__ void k_f32_to_f16(const float* __restrict__ in,
                             _Float16* __restrict__ out, int n) {
  int i = blockIdx.x * 256 + threadIdx.x;
  if (i < n) out[i] = (_Float16)in[i];
}

// conv weight [Cout, Cin, 4, 4] -> f16 [Mpad][Kpad], K = ci*16 + kh*4 + kw
__global__ void k_pack_w_conv(const float* __restrict__ w,
                              _Float16* __restrict__ out,
                              int Cout, int Cin, int Mpad, int Kpad) {
  int i = blockIdx.x * 256 + threadIdx.x;
  if (i >= Mpad * Kpad) return;
  int m = i / Kpad, k = i - m * Kpad;
  float v = 0.f;
  if (m < Cout && k < Cin * 16) v = w[m * Cin * 16 + k];
  out[i] = (_Float16)v;
}

// deconv weight (torch [Cin, Cout, 4, 4]) -> flipped OIHW, f16 [Mpad][Kpad]
__global__ void k_pack_w_deconv(const float* __restrict__ w,
                                _Float16* __restrict__ out,
                                int Cout, int Cin, int Mpad, int Kpad) {
  int i = blockIdx.x * 256 + threadIdx.x;
  if (i >= Mpad * Kpad) return;
  int m = i / Kpad, k = i - m * Kpad;
  float v = 0.f;
  if (m < Cout && k < Cin * 16) {
    int ci = k >> 4, r = k & 15, kh = r >> 2, kw = r & 3;
    v = w[((ci * Cout + m) * 4 + (3 - kh)) * 4 + (3 - kw)];
  }
  out[i] = (_Float16)v;
}

// codebook -> f16 + per-row squared norms
__global__ void k_pack_emb(const float* __restrict__ emb,
                           _Float16* __restrict__ out,
                           float* __restrict__ enorm) {
  int j = blockIdx.x * 64 + threadIdx.x;
  if (j >= 512) return;
  float s = 0.f;
  for (int k = 0; k < 32; ++k) {
    float v = emb[j * 32 + k];
    s += v * v;
    out[j * 32 + k] = (_Float16)v;
  }
  enorm[j] = s;
}

// ---------------------------------------------------------------------------
// Implicit-GEMM conv / transposed-conv with WMMA.
//   mode    : 0 = conv k4 s2 p1, 1 = convT k4 s2 p1 (weights pre-flipped)
//   actmode : 0 = none, 1 = relu, 2 = sigmoid + recon-loss accumulation
// All of Hi, Wi, Ho, Wo, HoWo are powers of two.
// ---------------------------------------------------------------------------
__global__ __launch_bounds__(256)
void k_wmma_conv(const _Float16* __restrict__ act,
                 const _Float16* __restrict__ wpack,
                 const float* __restrict__ bias,
                 _Float16* __restrict__ out16,
                 float* __restrict__ out32,
                 const float* __restrict__ xref,
                 float* __restrict__ loss_acc,
                 int Cin, int Hi, int Wi,
                 int Cout, int Ho, int Wo,
                 int Kpad, int mode, int actmode) {
  __shared__ __align__(32) _Float16 Bt[TILE_N][KC];   // im2col [n][k]

  const int t     = threadIdx.x;
  const int lane  = t & 31;
  const int wave  = t >> 5;
  const int half  = lane >> 4;
  const int ml    = lane & 15;
  const int nbase = blockIdx.x * TILE_N;
  const int mbase = blockIdx.y * TILE_M;

  const int HiWi = Hi * Wi;
  const int HoWo = Ho * Wo;
  const int woS  = __builtin_ctz((unsigned)Wo);
  const int hwS  = __builtin_ctz((unsigned)HoWo);

  // ---- per-thread gather geometry (chunk-invariant) -----------------------
  const int gn   = t >> 1;          // n within tile (0..127)
  const int kh16 = (t & 1) * 16;    // which K half this thread fills
  const int n_g  = nbase + gn;
  const int b_   = n_g >> hwS;
  const int hw   = n_g & (HoWo - 1);
  const int ho   = hw >> woS;
  const int wo   = hw & (Wo - 1);
  const int baseB = b_ * (Cin * HiWi);

  int taps[16];
  v16h mw;                                   // 0/1 f16 spatial mask
#pragma unroll
  for (int i = 0; i < 16; ++i) {
    int kh = i >> 2, kw = i & 3;
    int hi, wi;
    bool ok;
    if (mode == 0) {                         // conv, stride 2, pad 1
      hi = 2 * ho - 1 + kh;
      wi = 2 * wo - 1 + kw;
      ok = ((unsigned)hi < (unsigned)Hi) & ((unsigned)wi < (unsigned)Wi);
    } else {                                 // convT as dilated conv, pad 2
      int h2 = ho + kh - 2, w2 = wo + kw - 2;
      ok = (h2 >= 0) & (w2 >= 0) & ((h2 & 1) == 0) & ((w2 & 1) == 0);
      hi = h2 >> 1;
      wi = w2 >> 1;
      ok = ok & (hi < Hi) & (wi < Wi);
    }
    taps[i] = ok ? hi * Wi + wi : 0;         // clamped -> always in-bounds
    mw[i]   = ok ? (_Float16)1.f : (_Float16)0.f;
  }

  // ---- weight fragment row pointers (direct global, no LDS stage) ---------
  const _Float16* wrow0 = wpack + (mbase + ml) * Kpad;
  const _Float16* wrow1 = wpack + (mbase + 16 + ml) * Kpad;

  v8f c0 = {}, c1 = {};

  // ---- software-pipelined gather: issue chunk 0 ----
  // Channel index is clamped: K columns beyond Cin*16 multiply against
  // zero-padded weight rows, so their (finite) B values are don't-care.
  const int cimax = Cin - 1;
  v16h tv;
  {
    const int ci = (t & 1) < cimax ? (t & 1) : cimax;
    const unsigned cbase = (unsigned)(baseB + ci * HiWi);
#pragma unroll
    for (int i = 0; i < 16; ++i) tv[i] = act[cbase + (unsigned)taps[i]];
  }

  const int nchunks = Kpad >> 5;
  for (int kc = 0; kc < nchunks; ++kc) {
    // consume chunk kc: 8x v_pk_mul_f16 mask, one v16h LDS store
    v16h vals = tv * mw;
    *(v16h*)(&Bt[gn][kh16]) = vals;

    // issue chunk kc+1 (latency hides behind barrier + WMMA)
    if (kc + 1 < nchunks) {
      int ci = (kc + 1) * 2 + (t & 1);
      ci = ci < cimax ? ci : cimax;
      const unsigned cbase = (unsigned)(baseB + ci * HiWi);
#pragma unroll
      for (int i = 0; i < 16; ++i) tv[i] = act[cbase + (unsigned)taps[i]];
    }
    __syncthreads();

    // ---- fragments in contiguous k = 16*half + e order (A and B match), then 2x WMMA ----
    const int kb = kc * KC;
    const v8h a0lo = *(const v8h*)(wrow0 + kb + 16 * half);
    const v8h a0hi = *(const v8h*)(wrow0 + kb + 16 * half + 8);
    const v8h a1lo = *(const v8h*)(wrow1 + kb + 16 * half);
    const v8h a1hi = *(const v8h*)(wrow1 + kb + 16 * half + 8);
    v16h a0, a1, b;
#pragma unroll
    for (int e = 0; e < 8; ++e) {
      a0[e] = a0lo[e]; a0[e + 8] = a0hi[e];
      a1[e] = a1lo[e]; a1[e + 8] = a1hi[e];
    }
    const int ncol = wave * 16 + ml;
#pragma unroll
    for (int e = 0; e < 16; ++e)             // B: k = e + 16*half
      b[e] = Bt[ncol][e + 16 * half];
    c0 = __builtin_amdgcn_wmma_f32_16x16x32_f16(false, a0, false, b,
                                                (short)0, c0, false, false);
    c1 = __builtin_amdgcn_wmma_f32_16x16x32_f16(false, a1, false, b,
                                                (short)0, c1, false, false);
    __syncthreads();
  }

  // ---- epilogue: lane holds col n = lane%16, rows m = r + 8*half (+16) ----
  const int nl    = wave * 16 + ml;
  const int n_out = nbase + nl;
  const int bo    = n_out >> hwS;
  const int hwo   = n_out & (HoWo - 1);
  float lsum = 0.f;
#pragma unroll
  for (int r = 0; r < 16; ++r) {
    const int m = mbase + (r & 7) + 8 * half + (r < 8 ? 0 : 16);
    const float cv = (r < 8) ? c0[r & 7] : c1[r & 7];
    if (m < Cout) {
      float v = cv + bias[m];
      unsigned o = (unsigned)(bo * Cout + m) * (unsigned)HoWo + (unsigned)hwo;
      if (actmode == 2) {
        v = 1.f / (1.f + __expf(-v));
        out32[o] = v;
        float d = v - xref[o];
        lsum += d * d;
      } else {
        if (actmode == 1) v = v > 0.f ? v : 0.f;
        if (out16) out16[o] = (_Float16)v;
        if (out32) out32[o] = v;
      }
    }
  }
  if (actmode == 2) {
    for (int off = 16; off > 0; off >>= 1)
      lsum += __shfl_down(lsum, off, 32);
    if (lane == 0) atomicAdd(loss_acc, lsum);
  }
}

// ---------------------------------------------------------------------------
// VQ: per wave, 16 pixels. dot(z, emb^T) via one WMMA per 16-code tile
// (K = LATENT = 32 exactly). argmin of ||e||^2 - 2*dot, straight-through
// forward value z_q = emb[argmin], vq_loss = 1.25 * mse(q, z).
// ---------------------------------------------------------------------------
__global__ __launch_bounds__(32)
void k_vq(const _Float16* __restrict__ zf16,
          const float* __restrict__ zf32,
          const _Float16* __restrict__ embf16,
          const float* __restrict__ embf32,
          const float* __restrict__ enorm,
          _Float16* __restrict__ zqf16,
          float* __restrict__ vq_acc) {
  const int lane = threadIdx.x;
  const int half = lane >> 4;
  const int nb   = blockIdx.x * 16;

  // B fragment: columns = pixels, K = 32 latent dims (built once)
  v16h bz;
  {
    unsigned n = nb + (lane & 15);
    unsigned b_ = n >> 8, hw = n & 255;
#pragma unroll
    for (int e = 0; e < 16; ++e) {
      unsigned k = e + 16 * half;
      bz[e] = zf16[((b_ * 32 + k) << 8) + hw];
    }
  }

  float best = 3.0e38f;
  int bestj = 0;
  const int m = lane & 15;
  for (int jt = 0; jt < 32; ++jt) {          // 512 codes / 16
    const _Float16* row = embf16 + (jt * 16 + m) * 32;
    const v8h alo = *(const v8h*)(row + 16 * half);
    const v8h ahi = *(const v8h*)(row + 16 * half + 8);
    v16h a;
#pragma unroll
    for (int e = 0; e < 8; ++e) { a[e] = alo[e]; a[e + 8] = ahi[e]; }
    v8f c = {};
    c = __builtin_amdgcn_wmma_f32_16x16x32_f16(false, a, false, bz,
                                               (short)0, c, false, false);
#pragma unroll
    for (int r = 0; r < 8; ++r) {
      int j = jt * 16 + r + 8 * half;
      float d = enorm[j] - 2.f * c[r];
      if (d < best) { best = d; bestj = j; }
    }
  }
  // combine the two half-lanes that share a column
  float obest = __shfl_xor(best, 16, 32);
  int   oj    = __shfl_xor(bestj, 16, 32);
  if (obest < best || (obest == best && oj < bestj)) { best = obest; bestj = oj; }

  if (lane < 16) {
    unsigned n = nb + lane;
    unsigned b_ = n >> 8, hw = n & 255;
    float s = 0.f;
#pragma unroll
    for (int k = 0; k < 32; ++k) {
      float q = embf32[bestj * 32 + k];
      unsigned o = ((b_ * 32 + k) << 8) + hw;
      float z = zf32[o];
      float d = q - z;
      s += d * d;
      zqf16[o] = (_Float16)q;
    }
    atomicAdd(vq_acc, s);
  }
}

__global__ void k_finalize(const float* __restrict__ acc,
                           float* __restrict__ out) {
  if (threadIdx.x == 0) {
    out[6291456] = acc[0] / 6291456.f;               // recon_loss
    out[6291457] = 1.25f * acc[1] / 1048576.f;       // vq_loss (q+0.25*e)
  }
}

// ---------------------------------------------------------------------------
extern "C" void kernel_launch(void* const* d_in, const int* in_sizes, int n_in,
                              void* d_out, int out_size, void* d_ws,
                              size_t ws_size, hipStream_t stream) {
  (void)in_sizes; (void)n_in; (void)out_size; (void)ws_size;
  const float* x   = (const float*)d_in[0];
  const float* w1  = (const float*)d_in[1];
  const float* b1  = (const float*)d_in[2];
  const float* w2  = (const float*)d_in[3];
  const float* b2  = (const float*)d_in[4];
  const float* w3  = (const float*)d_in[5];
  const float* b3  = (const float*)d_in[6];
  const float* emb = (const float*)d_in[7];
  const float* d1  = (const float*)d_in[8];
  const float* db1 = (const float*)d_in[9];
  const float* d2  = (const float*)d_in[10];
  const float* db2 = (const float*)d_in[11];
  const float* d3  = (const float*)d_in[12];
  const float* db3 = (const float*)d_in[13];
  float* out = (float*)d_out;

  // ---- workspace carve-out (256B aligned)
  char* base = (char*)d_ws;
  size_t off = 0;
  auto carve = [&](size_t bytes) -> void* {
    off = (off + 255) & ~(size_t)255;
    void* p = base + off;
    off += bytes;
    return p;
  };
  float*    acc    = (float*)carve(8 * sizeof(float));        // [0]=recon [1]=vq
  _Float16* xf16   = (_Float16*)carve((size_t)6291456 * 2);   // 128x3x128x128
  _Float16* h1     = (_Float16*)carve((size_t)16777216 * 2);  // 128x32x64x64
  _Float16* h2     = (_Float16*)carve((size_t)8388608 * 2);   // 128x64x32x32
  _Float16* zf16   = (_Float16*)carve((size_t)1048576 * 2);   // 128x32x16x16
  float*    zf32   = (float*)carve((size_t)1048576 * 4);
  _Float16* zq     = (_Float16*)carve((size_t)1048576 * 2);
  _Float16* g1     = (_Float16*)carve((size_t)8388608 * 2);   // 128x64x32x32
  _Float16* g2     = (_Float16*)carve((size_t)16777216 * 2);  // 128x32x64x64
  _Float16* w1p    = (_Float16*)carve((size_t)32 * 64 * 2);
  _Float16* w2p    = (_Float16*)carve((size_t)64 * 512 * 2);
  _Float16* w3p    = (_Float16*)carve((size_t)32 * 1024 * 2);
  _Float16* d1p    = (_Float16*)carve((size_t)64 * 512 * 2);
  _Float16* d2p    = (_Float16*)carve((size_t)32 * 1024 * 2);
  _Float16* d3p    = (_Float16*)carve((size_t)32 * 512 * 2);
  _Float16* embf16 = (_Float16*)carve((size_t)512 * 32 * 2);
  float*    enorm  = (float*)carve((size_t)512 * 4);

  hipMemsetAsync(acc, 0, 8 * sizeof(float), stream);

  // ---- packing
  k_f32_to_f16<<<(6291456 + 255) / 256, 256, 0, stream>>>(x, xf16, 6291456);
  k_pack_w_conv<<<(32 * 64 + 255) / 256, 256, 0, stream>>>(w1, w1p, 32, 3, 32, 64);
  k_pack_w_conv<<<(64 * 512 + 255) / 256, 256, 0, stream>>>(w2, w2p, 64, 32, 64, 512);
  k_pack_w_conv<<<(32 * 1024 + 255) / 256, 256, 0, stream>>>(w3, w3p, 32, 64, 32, 1024);
  k_pack_w_deconv<<<(64 * 512 + 255) / 256, 256, 0, stream>>>(d1, d1p, 64, 32, 64, 512);
  k_pack_w_deconv<<<(32 * 1024 + 255) / 256, 256, 0, stream>>>(d2, d2p, 32, 64, 32, 1024);
  k_pack_w_deconv<<<(32 * 512 + 255) / 256, 256, 0, stream>>>(d3, d3p, 3, 32, 32, 512);
  k_pack_emb<<<8, 64, 0, stream>>>(emb, embf16, enorm);

  // ---- encoder (grid.x = N/128, grid.y = Mpad/32)
  k_wmma_conv<<<dim3(524288 / TILE_N, 1), 256, 0, stream>>>(
      xf16, w1p, b1, h1, nullptr, nullptr, nullptr,
      3, 128, 128, 32, 64, 64, 64, 0, 1);
  k_wmma_conv<<<dim3(131072 / TILE_N, 2), 256, 0, stream>>>(
      h1, w2p, b2, h2, nullptr, nullptr, nullptr,
      32, 64, 64, 64, 32, 32, 512, 0, 1);
  k_wmma_conv<<<dim3(32768 / TILE_N, 1), 256, 0, stream>>>(
      h2, w3p, b3, zf16, zf32, nullptr, nullptr,
      64, 32, 32, 32, 16, 16, 1024, 0, 0);

  // ---- vector quantization (32768 pixels, 16 per wave)
  k_vq<<<32768 / 16, 32, 0, stream>>>(zf16, zf32, embf16, emb, enorm, zq, acc + 1);

  // ---- decoder
  k_wmma_conv<<<dim3(131072 / TILE_N, 2), 256, 0, stream>>>(
      zq, d1p, db1, g1, nullptr, nullptr, nullptr,
      32, 16, 16, 64, 32, 32, 512, 1, 1);
  k_wmma_conv<<<dim3(524288 / TILE_N, 1), 256, 0, stream>>>(
      g1, d2p, db2, g2, nullptr, nullptr, nullptr,
      64, 32, 32, 32, 64, 64, 1024, 1, 1);
  k_wmma_conv<<<dim3(2097152 / TILE_N, 1), 256, 0, stream>>>(
      g2, d3p, db3, nullptr, out, x, acc,
      32, 64, 64, 3, 128, 128, 512, 1, 2);

  k_finalize<<<1, 1, 0, stream>>>(acc, out);
}
